// TaskAlignedAssignerLayer_80564996538882
// MI455X (gfx1250) — matrix operand order repaired and checked by reading, not compiled, not run on hardware
//
#include <hip/hip_runtime.h>
#include <hip/hip_bf16.h>
#include <cstdint>

// ---------------- problem constants ----------------
#define Bn   32
#define Mn   64
#define Kn   8400
#define Cn   80
#define TOPK 15
#define EPSf 1e-7f
#define KT   (Kn / 16)   // 525 k-tiles
#define MT   (Mn / 16)   // 4 m-tiles

typedef __attribute__((ext_vector_type(16))) _Float16 v16h;
typedef __attribute__((ext_vector_type(8)))  float    v8f;

// pack 16 consecutive floats (scaled by sel) into a v16h B-matrix fragment
__device__ __forceinline__ v16h pack16(float4 q0, float4 q1, float4 q2, float4 q3,
                                       float sel)
{
    v16h r;
    r[0]  = (_Float16)(q0.x * sel); r[1]  = (_Float16)(q0.y * sel);
    r[2]  = (_Float16)(q0.z * sel); r[3]  = (_Float16)(q0.w * sel);
    r[4]  = (_Float16)(q1.x * sel); r[5]  = (_Float16)(q1.y * sel);
    r[6]  = (_Float16)(q1.z * sel); r[7]  = (_Float16)(q1.w * sel);
    r[8]  = (_Float16)(q2.x * sel); r[9]  = (_Float16)(q2.y * sel);
    r[10] = (_Float16)(q2.z * sel); r[11] = (_Float16)(q2.w * sel);
    r[12] = (_Float16)(q3.x * sel); r[13] = (_Float16)(q3.y * sel);
    r[14] = (_Float16)(q3.z * sel); r[15] = (_Float16)(q3.w * sel);
    return r;
}

// A-matrix fragment: onehot(lab) for class chunk starting at c0
// 16-bit A 16x32 layout (contiguous, matches B): element j -> K=j+16*hi
__device__ __forceinline__ v16h onehotA(int c0, int hi, int lab)
{
    v16h a;
#pragma unroll
    for (int j = 0; j < 16; ++j) {
        int v  = j >> 1, h = j & 1;
        int ka = j + 16 * hi; (void)v; (void)h;
        a[j] = ((c0 + ka) == lab) ? (_Float16)1.0f : (_Float16)0.0f;
    }
    return a;
}

// =====================================================================
// Kernel 1: per 16x16 (m,k) tile per wave:
//   ts(m,k) = sum_c onehot(label_m, c) * scores(k, c)   via v_wmma_f32_16x16x32_f16
//   then each lane computes CIoU for its 8 accumulator elements.
// Writes overlaps (output) and am = ts * ov^6 (workspace).
// =====================================================================
__global__ __launch_bounds__(256) void k_am_ov(
    const float* __restrict__ points, const float* __restrict__ gtb,
    const int*   __restrict__ glab,   const float* __restrict__ pbox,
    const float* __restrict__ pscore, const int*   __restrict__ gmask,
    float* __restrict__ out_ov, float* __restrict__ ws_am)
{
    int wave = threadIdx.x >> 5;
    int lane = threadIdx.x & 31;
    int tile = blockIdx.x * 8 + wave;          // grid sized exactly: no guard -> EXEC all 1s
    int b   = tile / (MT * KT);
    int rem = tile % (MT * KT);
    int mt  = rem / KT, kt = rem % KT;
    int m0  = mt * 16, k0 = kt * 16;
    int r16 = lane & 15, hi = lane >> 4;

    int lab = glab[b * Mn + m0 + r16];         // A-matrix row label (both halves: M=lane&15)
    int k   = k0 + r16;                        // B-matrix column (N = lane&15)
    const float* srow = pscore + ((size_t)b * Kn + k) * (size_t)Cn;

    v8f acc = {0.f,0.f,0.f,0.f,0.f,0.f,0.f,0.f};

    // chunks c0 = 0, 32: all 32 classes valid -> unconditional b128 loads
#pragma unroll
    for (int c0 = 0; c0 < 64; c0 += 32) {
        const float4* qp = reinterpret_cast<const float4*>(srow + c0 + 16 * hi);
        float4 q0 = qp[0], q1 = qp[1], q2 = qp[2], q3 = qp[3];
        v16h Bmat = pack16(q0, q1, q2, q3, 1.0f);
        v16h Amat = onehotA(c0, hi, lab);
        acc = __builtin_amdgcn_wmma_f32_16x16x32_f16(
            false, Amat, false, Bmat, (short)0, acc, false, false);
    }
    // chunk c0 = 64: classes 64..79 valid (hi==0 half); hi==1 half (80..95) is
    // zero-padding. Both halves load the clamped, always-valid range srow+64..79
    // and the hi==1 half is zeroed branch-free via sel — EXEC never changes.
    {
        const float4* qp = reinterpret_cast<const float4*>(srow + 64);
        float4 q0 = qp[0], q1 = qp[1], q2 = qp[2], q3 = qp[3];
        float sel = (hi == 0) ? 1.0f : 0.0f;
        v16h Bmat = pack16(q0, q1, q2, q3, sel);
        v16h Amat = onehotA(64, hi, lab);      // ca in 80..95 never matches lab for pad
        acc = __builtin_amdgcn_wmma_f32_16x16x32_f16(
            false, Amat, false, Bmat, (short)0, acc, false, false);
    }

    // C/D layout: VGPR r holds (M = m0 + r + 8*hi, N = k0 + (lane&15))
    const float2 pt = reinterpret_cast<const float2*>(points)[k];
    const float4 pb = reinterpret_cast<const float4*>(pbox)[(size_t)b * Kn + k];
#pragma unroll
    for (int r = 0; r < 8; ++r) {
        int m = m0 + 8 * hi + r;
        float4 gb = reinterpret_cast<const float4*>(gtb)[b * Mn + m];
        bool inb = (gb.x - pt.x < 1e-6f) && (gb.y - pt.y < 1e-6f) &&
                   (pt.x - gb.z < 1e-6f) && (pt.y - gb.w < 1e-6f);
        float posf = (inb && (gmask[b * Mn + m] != 0)) ? 1.f : 0.f;
        // CIoU(gt, pred)
        float w1 = gb.z - gb.x, h1 = gb.w - gb.y;
        float w2 = pb.z - pb.x, h2 = pb.w - pb.y;
        float iw = fmaxf(fminf(gb.z, pb.z) - fmaxf(gb.x, pb.x), 0.f);
        float ih = fmaxf(fminf(gb.w, pb.w) - fmaxf(gb.y, pb.y), 0.f);
        float inter = iw * ih;
        float uni   = w1 * h1 + w2 * h2 - inter + EPSf;
        float iou   = inter / uni;
        float cw = fmaxf(gb.z, pb.z) - fminf(gb.x, pb.x);
        float ch = fmaxf(gb.w, pb.w) - fminf(gb.y, pb.y);
        float c2 = cw * cw + ch * ch + EPSf;
        float dx = pb.x + pb.z - gb.x - gb.z;
        float dy = pb.y + pb.w - gb.y - gb.w;
        float rho2 = (dx * dx + dy * dy) * 0.25f;
        float t  = atanf(w2 / (h2 + EPSf)) - atanf(w1 / (h1 + EPSf));
        float vv = 0.4052847345693511f * t * t;       // 4/pi^2
        float alpha = vv / (vv - iou + (1.f + EPSf));
        float ci = iou - (rho2 / c2 + alpha * vv);

        float ov  = fmaxf(ci, 0.f) * posf;
        float ts  = acc[r] * posf;
        float ov2 = ov * ov;
        float am  = ts * (ov2 * ov2 * ov2);           // CLS_POW=1, IOU_POW=6
        size_t idx = ((size_t)(b * Mn + m)) * Kn + k;
        out_ov[idx] = ov;
        ws_am[idx]  = am;
    }
}

// =====================================================================
// Kernel 2: top-15 per (b,m) row over K=8400; row staged in LDS.
// assign[idx] = 1 iff idx picked AND value > 1e-9 (masked picks add 0 at idx 0).
// =====================================================================
__global__ __launch_bounds__(256) void k_topk(
    const float* __restrict__ ws_am, float* __restrict__ out_assign)
{
    __shared__ float s_am[Kn];
    __shared__ float smV[256];
    __shared__ int   smI[256];
    int tid = threadIdx.x;
    int row = blockIdx.x;                       // b*Mn + m
    const float* ar  = ws_am      + (size_t)row * Kn;
    float*       asr = out_assign + (size_t)row * Kn;
    for (int i = tid; i < Kn; i += 256) { s_am[i] = ar[i]; asr[i] = 0.f; }
    __syncthreads();
    for (int t = 0; t < TOPK; ++t) {
        float lv = -2.f; int li = Kn;
        for (int i = tid; i < Kn; i += 256) {   // ascending: strict > keeps lowest index
            float v = s_am[i];
            if (v > lv) { lv = v; li = i; }
        }
        smV[tid] = lv; smI[tid] = li;
        __syncthreads();
        for (int s = 128; s > 0; s >>= 1) {
            if (tid < s) {
                float v2 = smV[tid + s]; int i2 = smI[tid + s];
                if (v2 > smV[tid] || (v2 == smV[tid] && i2 < smI[tid])) {
                    smV[tid] = v2; smI[tid] = i2;
                }
            }
            __syncthreads();
        }
        if (tid == 0) {
            int wi = smI[0]; float wv = smV[0];
            if (wv > 1e-9f) asr[wi] = 1.f;
            s_am[wi] = -1.f;                    // exclude from later picks
        }
        __syncthreads();
    }
}

// =====================================================================
// Kernel 3: flag = (max over (b,k) of sum_m assign) > 1   (lax.cond predicate)
// =====================================================================
__global__ void k_flag(const float* __restrict__ assign, int* __restrict__ flag)
{
    int tid = blockIdx.x * blockDim.x + threadIdx.x;
    if (tid >= Bn * Kn) return;
    int b = tid / Kn, k = tid % Kn;
    const float* base = assign + ((size_t)b * Mn) * Kn + k;
    int s = 0;
    for (int m = 0; m < Mn; ++m) s += (base[(size_t)m * Kn] > 0.5f);
    if (s > 1) atomicOr(flag, 1);
}

// =====================================================================
// Kernel 4: per (b,k): conditional overlap-argmax filter, fg, tbi,
//           tgt_bboxes, tgt_label.
// =====================================================================
__global__ void k_filter(
    const float* __restrict__ ov, const int* __restrict__ glab,
    const float* __restrict__ gtb,
    float* __restrict__ assign, float* __restrict__ out_tgb,
    float* __restrict__ out_fg, int* __restrict__ ws_lab,
    int* __restrict__ ws_fg, const int* __restrict__ flag)
{
    int tid = blockIdx.x * blockDim.x + threadIdx.x;
    if (tid >= Bn * Kn) return;
    int b = tid / Kn, k = tid % Kn;
    size_t base = ((size_t)b * Mn) * Kn + k;
    unsigned long long abits = 0ull;
    float bestov = -1e30f; int mi = 0;
    for (int m = 0; m < Mn; ++m) {
        float a = assign[base + (size_t)m * Kn];
        float o = ov[base + (size_t)m * Kn];
        if (a > 0.5f) abits |= (1ull << m);
        if (o > bestov) { bestov = o; mi = m; }      // first-max, like jnp.argmax
    }
    if (*flag) {
        unsigned long long keep = abits & (1ull << mi);
        if (keep != abits) {
            for (int m = 0; m < Mn; ++m)
                assign[base + (size_t)m * Kn] = (float)((keep >> m) & 1ull);
        }
        abits = keep;
    }
    int fg  = __builtin_popcountll(abits);
    int tbi = abits ? __builtin_ctzll(abits) : 0;    // argmax of 0/1 -> first 1, else 0
    int lab = glab[b * Mn + tbi];
    reinterpret_cast<float4*>(out_tgb)[tid] =
        reinterpret_cast<const float4*>(gtb)[b * Mn + tbi];
    out_fg[tid] = (float)fg;
    ws_lab[tid] = lab;
    ws_fg[tid]  = fg;
}

// =====================================================================
// Kernel 5: per (b,m): rowmaxA = max_k am*assign ; rowmaxO = max_k overlaps
// =====================================================================
__global__ __launch_bounds__(256) void k_rowmax(
    const float* __restrict__ am, const float* __restrict__ assign,
    const float* __restrict__ ov, float* __restrict__ rmA, float* __restrict__ rmO)
{
    __shared__ float sA[256], sO[256];
    int tid = threadIdx.x;
    int row = blockIdx.x;
    size_t base = (size_t)row * Kn;
    float mA = 0.f, mO = 0.f;                        // both quantities are >= 0
    for (int i = tid; i < Kn; i += 256) {
        mA = fmaxf(mA, am[base + i] * assign[base + i]);
        mO = fmaxf(mO, ov[base + i]);
    }
    sA[tid] = mA; sO[tid] = mO;
    __syncthreads();
    for (int s = 128; s > 0; s >>= 1) {
        if (tid < s) {
            sA[tid] = fmaxf(sA[tid], sA[tid + s]);
            sO[tid] = fmaxf(sO[tid], sO[tid + s]);
        }
        __syncthreads();
    }
    if (tid == 0) { rmA[row] = sA[0]; rmO[row] = sO[0]; }
}

// =====================================================================
// Kernel 6: per (b,k): factor = max_m am*assign/(rowmaxA+1e-9)*rowmaxO
// =====================================================================
__global__ void k_factor(
    const float* __restrict__ am, const float* __restrict__ assign,
    const float* __restrict__ rmA, const float* __restrict__ rmO,
    float* __restrict__ factor)
{
    int tid = blockIdx.x * blockDim.x + threadIdx.x;
    if (tid >= Bn * Kn) return;
    int b = tid / Kn, k = tid % Kn;
    size_t base = ((size_t)b * Mn) * Kn + k;
    float f = 0.f;                                   // am_n >= 0
    for (int m = 0; m < Mn; ++m) {
        int bm = b * Mn + m;
        float a = am[base + (size_t)m * Kn] * assign[base + (size_t)m * Kn];
        f = fmaxf(f, a / (rmA[bm] + 1e-9f) * rmO[bm]);
    }
    factor[tid] = f;
}

// =====================================================================
// Kernel 7: tso[b,k,c] = onehot(label)[c] * fg * factor
// =====================================================================
__global__ void k_tso(
    const int* __restrict__ ws_lab, const int* __restrict__ ws_fg,
    const float* __restrict__ factor, float* __restrict__ out_tso)
{
    size_t tid = (size_t)blockIdx.x * blockDim.x + threadIdx.x;
    size_t tot = (size_t)Bn * Kn * Cn;
    if (tid >= tot) return;
    int    c  = (int)(tid % Cn);
    size_t bk = tid / Cn;
    float v = (c == ws_lab[bk] && ws_fg[bk] != 0) ? factor[bk] : 0.f;
    out_tso[tid] = v;
}

// =====================================================================
extern "C" void kernel_launch(void* const* d_in, const int* in_sizes, int n_in,
                              void* d_out, int out_size, void* d_ws, size_t ws_size,
                              hipStream_t stream)
{
    const float* points = (const float*)d_in[0];   // (K,2)
    const float* gtb    = (const float*)d_in[1];   // (B,M,4)
    const int*   glab   = (const int*)  d_in[2];   // (B,M,1)
    const float* pbox   = (const float*)d_in[3];   // (B,K,4)
    const float* pscore = (const float*)d_in[4];   // (B,K,80)
    const int*   gmask  = (const int*)  d_in[5];   // (B,M)

    float* out        = (float*)d_out;
    float* out_tso    = out;
    float* out_tgb    = out_tso    + (size_t)Bn * Kn * Cn;
    float* out_assign = out_tgb    + (size_t)Bn * Kn * 4;
    float* out_ov     = out_assign + (size_t)Bn * Mn * Kn;
    float* out_fg     = out_ov     + (size_t)Bn * Mn * Kn;

    float* ws_am  = (float*)d_ws;                           // B*M*K
    float* rmA    = ws_am  + (size_t)Bn * Mn * Kn;          // B*M
    float* rmO    = rmA    + (size_t)Bn * Mn;               // B*M
    float* factor = rmO    + (size_t)Bn * Mn;               // B*K
    int*   ws_lab = (int*)(factor + (size_t)Bn * Kn);       // B*K
    int*   ws_fg  = ws_lab + (size_t)Bn * Kn;               // B*K
    int*   flag   = ws_fg  + (size_t)Bn * Kn;               // 1

    hipMemsetAsync(flag, 0, sizeof(int), stream);

    int tiles = Bn * MT * KT;                  // 67200, exactly 8 waves/block
    k_am_ov<<<tiles / 8, 256, 0, stream>>>(points, gtb, glab, pbox, pscore,
                                           gmask, out_ov, ws_am);
    k_topk<<<Bn * Mn, 256, 0, stream>>>(ws_am, out_assign);

    int bk = Bn * Kn;
    k_flag<<<(bk + 255) / 256, 256, 0, stream>>>(out_assign, flag);
    k_filter<<<(bk + 255) / 256, 256, 0, stream>>>(out_ov, glab, gtb, out_assign,
                                                   out_tgb, out_fg, ws_lab, ws_fg,
                                                   flag);
    k_rowmax<<<Bn * Mn, 256, 0, stream>>>(ws_am, out_assign, out_ov, rmA, rmO);
    k_factor<<<(bk + 255) / 256, 256, 0, stream>>>(ws_am, out_assign, rmA, rmO,
                                                   factor);
    size_t tot = (size_t)Bn * Kn * Cn;
    k_tso<<<(int)((tot + 255) / 256), 256, 0, stream>>>(ws_lab, ws_fg, factor,
                                                        out_tso);
}
